// DriftAwareGraphEncoder_84834194030659
// MI455X (gfx1250) — matrix operand order repaired and checked by reading, not compiled, not run on hardware
//
#include <hip/hip_runtime.h>
#include <hip/hip_bf16.h>

// ---------------------------------------------------------------------------
// DriftAwareGraphEncoder for MI455X (gfx1250, wave32, WMMA)
//
// Roofline: GEMMs total ~11 GFLOP -> trivial at f16 WMMA rates; the edge
// aggregation (650k edges x 256 ch gather + f32 atomic scatter, node arrays
// 51MB < 192MB L2) is the bound. GEMMs use v_wmma_f32_16x16x32_f16 with
// vectorized LDS fragment loads (ds_load_b128) and a transposed B tile so the
// matrix pipe is not starved by scalar ds_load_u16 gathers.
// ---------------------------------------------------------------------------

typedef _Float16 h16 __attribute__((ext_vector_type(16)));
typedef _Float16 h8v __attribute__((ext_vector_type(8)));
typedef float f32x8 __attribute__((ext_vector_type(8)));

struct alignas(16) H8 { _Float16 v[8]; };
struct alignas(8)  H4 { _Float16 v[4]; };

// ---------------- GEMM: C[M,Nc] = A[M,K](f16) @ B[K,Nc](f16), f32 accum ----
// Block: 256 threads = 8 wave32s; block tile 128x32, K-step 32.
// Wave w owns rows m0+16w..+15 and both 16-col tiles (2 WMMAs / K-step,
// shared A fragment). A fragment: 2x ds_load_b128 (contiguous k = 16*(lane/16)+s,
// same order as B). B tile stored transposed [n][k] so the B fragment is
// also 2x ds_load_b128 per tile.
__global__ __launch_bounds__(256) void gemm_f16_wmma(
    const _Float16* __restrict__ A, const _Float16* __restrict__ B,
    float* __restrict__ C, int M, int K, int Nc) {
  __shared__ _Float16 As[128 * 32];   // [row][k]  8 KB
  __shared__ _Float16 Bt[32 * 32];    // [n][k] transposed, 2 KB
  const int tid  = threadIdx.x;
  const int wave = tid >> 5;          // 0..7 (wave32)
  const int lane = tid & 31;
  const int m0   = blockIdx.x * 128;
  const int bn0  = blockIdx.y * 32;

  f32x8 acc0 = {}, acc1 = {};

  const int arow = tid >> 1;          // 0..127
  const int acol = (tid & 1) * 16;    // 0 or 16
  const int brow = tid >> 3;          // 0..31  (k)
  const int bcol = (tid & 7) * 4;     // 0..28  (n)

  const int ml = lane & 15, hi = lane >> 4;
  const int kb = hi * 8, kbb = hi * 16;
  const _Float16* arp  = As + (wave * 16 + ml) * 32;
  const _Float16* brp0 = Bt + ml * 32 + kbb;
  const _Float16* brp1 = Bt + (16 + ml) * 32 + kbb;

  for (int k0 = 0; k0 < K; k0 += 32) {
    // ---- stage A tile (coalesced 16B loads, zero-pad rows >= M) ----
    H8 v0{}, v1{};
    const int gr = m0 + arow;
    if (gr < M) {
      const H8* p = (const H8*)(A + (size_t)gr * K + k0 + acol);
      v0 = p[0];
      v1 = p[1];
      if (k0 + 32 < K)
        __builtin_prefetch(A + (size_t)gr * K + k0 + 32 + acol, 0, 1);
    }
    *((H8*)(As + arow * 32 + acol))     = v0;
    *((H8*)(As + arow * 32 + acol + 8)) = v1;
    // ---- stage B tile transposed: coalesced 8B global load, 4x b16 store --
    {
      H4 bv = *((const H4*)(B + (size_t)(k0 + brow) * Nc + bn0 + bcol));
#pragma unroll
      for (int j = 0; j < 4; ++j) Bt[(bcol + j) * 32 + brow] = bv.v[j];
    }
    __syncthreads();

    // ---- vector fragment loads ----
    h8v alo = *(const h8v*)(arp + kbb);
    h8v ahi = *(const h8v*)(arp + kbb + 8);
    h16 af = __builtin_shufflevector(alo, ahi, 0, 1, 2, 3, 4, 5, 6, 7, 8, 9,
                                     10, 11, 12, 13, 14, 15);
    h8v b0l = *(const h8v*)(brp0);
    h8v b0h = *(const h8v*)(brp0 + 8);
    h16 bf0 = __builtin_shufflevector(b0l, b0h, 0, 1, 2, 3, 4, 5, 6, 7, 8, 9,
                                      10, 11, 12, 13, 14, 15);
    h8v b1l = *(const h8v*)(brp1);
    h8v b1h = *(const h8v*)(brp1 + 8);
    h16 bf1 = __builtin_shufflevector(b1l, b1h, 0, 1, 2, 3, 4, 5, 6, 7, 8, 9,
                                      10, 11, 12, 13, 14, 15);

    acc0 = __builtin_amdgcn_wmma_f32_16x16x32_f16(
        false, af, false, bf0, (short)0, acc0, false, false);
    acc1 = __builtin_amdgcn_wmma_f32_16x16x32_f16(
        false, af, false, bf1, (short)0, acc1, false, false);
    __syncthreads();
  }

  // ---- store D: VGPR r -> M = r + (lane>=16)*8, N = lane&15 ----
  const int colbase = bn0 + ml;
  if (m0 + 128 <= M) {   // fast path: no per-element guards
#pragma unroll
    for (int r = 0; r < 8; ++r) {
      const size_t row = (size_t)(m0 + wave * 16 + hi * 8 + r);
      C[row * Nc + colbase]      = acc0[r];
      C[row * Nc + colbase + 16] = acc1[r];
    }
  } else {
#pragma unroll
    for (int r = 0; r < 8; ++r) {
      const int row = m0 + wave * 16 + hi * 8 + r;
      if (row < M) {
        C[(size_t)row * Nc + colbase]      = acc0[r];
        C[(size_t)row * Nc + colbase + 16] = acc1[r];
      }
    }
  }
}

// ---------------- small elementwise / graph kernels ------------------------

__global__ void k_f32_to_f16(const float* __restrict__ in,
                             _Float16* __restrict__ out, long n) {
  long i = (long)blockIdx.x * blockDim.x + threadIdx.x;
  if (i < n) out[i] = (_Float16)in[i];
}

__global__ void k_zero_f32(float* __restrict__ p, long n) {
  long i = (long)blockIdx.x * blockDim.x + threadIdx.x;
  if (i < n) p[i] = 0.0f;
}

// s_src[n,h] = sum_c h[n,h*64+c]*a_src[h,c]; same for dst. One thread per (n,h).
__global__ void k_head_dots(const float* __restrict__ h,
                            const float* __restrict__ aS,
                            const float* __restrict__ aD,
                            float* __restrict__ sS, float* __restrict__ sD,
                            int N) {
  long t = (long)blockIdx.x * blockDim.x + threadIdx.x;
  if (t >= (long)N * 4) return;
  const int hd = (int)(t & 3);
  const float* hp = h + (t >> 2) * 256 + hd * 64;
  const float* as = aS + hd * 64;
  const float* ad = aD + hd * 64;
  float d1 = 0.f, d2 = 0.f;
#pragma unroll 4
  for (int i = 0; i < 64; ++i) {
    float v = hp[i];
    d1 += v * as[i];
    d2 += v * ad[i];
  }
  sS[t] = d1;
  sD[t] = d2;
}

// p_e = exp(leakyrelu(sS[src]+sD[dst])); z[dst] += p  (softmax denom).
// Max-subtraction omitted: softmax-invariant, logits are O(1) here.
__global__ void k_edge_scores(const int* __restrict__ ei, int E, int N,
                              const float* __restrict__ sS,
                              const float* __restrict__ sD,
                              float* __restrict__ p, float* __restrict__ z) {
  long e = (long)blockIdx.x * blockDim.x + threadIdx.x;
  if (e >= (long)E + N) return;
  int src, dst;
  if (e < E) { src = ei[e]; dst = ei[E + e]; }
  else       { src = dst = (int)(e - E); }
#pragma unroll
  for (int hd = 0; hd < 4; ++hd) {
    float v = sS[(size_t)src * 4 + hd] + sD[(size_t)dst * 4 + hd];
    v = v > 0.f ? v : 0.2f * v;     // LeakyReLU(0.2)
    float pe = expf(v);
    p[e * 4 + hd] = pe;
    atomicAdd(&z[(size_t)dst * 4 + hd], pe);
  }
}

// out[dst, c] += h[src, c] * alpha(e, head(c)); C=256. 8 edges per block,
// one wave per edge, lane-coalesced channels c = j*32+lane.
__global__ __launch_bounds__(256) void k_gat_aggregate(
    const int* __restrict__ ei, int E, int N, const float* __restrict__ p,
    const float* __restrict__ z, const float* __restrict__ h,
    float* __restrict__ out) {
  const int lane = threadIdx.x & 31;
  long e = (long)blockIdx.x * 8 + (threadIdx.x >> 5);
  if (e >= (long)E + N) return;
  int src, dst;
  if (e < E) { src = ei[e]; dst = ei[E + e]; }
  else       { src = dst = (int)(e - E); }
#pragma unroll
  for (int j = 0; j < 8; ++j) {
    const int c  = j * 32 + lane;
    const int hd = c >> 6;
    const float alpha = p[e * 4 + hd] / z[(size_t)dst * 4 + hd];
    atomicAdd(&out[(size_t)dst * 256 + c], h[(size_t)src * 256 + c] * alpha);
  }
}

// BatchNorm pass 1: per-feature partial sums over a 128-row slab.
__global__ __launch_bounds__(256) void k_bn_reduce(
    const float* __restrict__ h, float* __restrict__ sum,
    float* __restrict__ sumsq, int N) {
  const int f = threadIdx.x;            // 256 features
  const int r0 = blockIdx.x * 128;
  int r1 = r0 + 128;
  if (r1 > N) r1 = N;
  float s = 0.f, q = 0.f;
  for (int r = r0; r < r1; ++r) {
    float v = h[(size_t)r * 256 + f];
    s += v;
    q += v * v;
  }
  atomicAdd(&sum[f], s);
  atomicAdd(&sumsq[f], q);
}

__global__ void k_bn_finalize(const float* __restrict__ sum,
                              const float* __restrict__ sumsq,
                              const float* __restrict__ gamma,
                              const float* __restrict__ beta,
                              float* __restrict__ scale,
                              float* __restrict__ shift, float invN) {
  const int f = threadIdx.x;
  const float mu  = sum[f] * invN;
  const float var = sumsq[f] * invN - mu * mu;   // biased var, as jnp .var()
  const float sc  = rsqrtf(var + 1e-5f) * gamma[f];
  scale[f] = sc;
  shift[f] = beta[f] - mu * sc;
}

// hbn16 = f16(elu(h*scale + shift))
__global__ void k_bn_elu_f16(const float* __restrict__ h,
                             const float* __restrict__ scale,
                             const float* __restrict__ shift,
                             _Float16* __restrict__ out, long n) {
  long i = (long)blockIdx.x * blockDim.x + threadIdx.x;
  if (i >= n) return;
  const int f = (int)(i & 255);
  float v = h[i] * scale[f] + shift[f];
  v = v > 0.f ? v : (expf(v) - 1.f);
  out[i] = (_Float16)v;
}

// h_gat[n,c] = mean over 4 heads + b2[c]
__global__ void k_head_mean_bias(const float* __restrict__ agg,
                                 const float* __restrict__ b,
                                 float* __restrict__ out, int N) {
  long t = (long)blockIdx.x * blockDim.x + threadIdx.x;
  if (t >= (long)N * 64) return;
  const int c = (int)(t & 63);
  const float* a = agg + (t >> 6) * 256;
  out[t] = 0.25f * (a[c] + a[64 + c] + a[128 + c] + a[192 + c]) + b[c];
}

__global__ void k_deg_init(float* __restrict__ deg, int N) {
  int i = blockIdx.x * blockDim.x + threadIdx.x;
  if (i < N) deg[i] = 1.0f;   // self loop
}
__global__ void k_deg_accum(const int* __restrict__ ei, int E,
                            float* __restrict__ deg) {
  long e = (long)blockIdx.x * blockDim.x + threadIdx.x;
  if (e < E) atomicAdd(&deg[ei[E + e]], 1.0f);
}
__global__ void k_dinv(float* __restrict__ d, int N) {
  int i = blockIdx.x * blockDim.x + threadIdx.x;
  if (i < N) d[i] = rsqrtf(fmaxf(d[i], 1.0f));
}

// GCN: out[dst,c] += h[src,c] * dinv[src]*dinv[dst]; C=64, one wave per edge.
__global__ __launch_bounds__(256) void k_gcn_aggregate(
    const int* __restrict__ ei, int E, int N, const float* __restrict__ dinv,
    const float* __restrict__ h, float* __restrict__ out) {
  const int lane = threadIdx.x & 31;
  long e = (long)blockIdx.x * 8 + (threadIdx.x >> 5);
  if (e >= (long)E + N) return;
  int src, dst;
  if (e < E) { src = ei[e]; dst = ei[E + e]; }
  else       { src = dst = (int)(e - E); }
  const float norm = dinv[src] * dinv[dst];
#pragma unroll
  for (int j = 0; j < 2; ++j) {
    const int c = j * 32 + lane;
    atomicAdd(&out[(size_t)dst * 64 + c], h[(size_t)src * 64 + c] * norm);
  }
}

// g16 = f16(relu(in + bg1[c]))  (C=64)
__global__ void k_bias_relu_f16(const float* __restrict__ in,
                                const float* __restrict__ b,
                                _Float16* __restrict__ out, long n) {
  long i = (long)blockIdx.x * blockDim.x + threadIdx.x;
  if (i >= n) return;
  float v = in[i] + b[(int)(i & 63)];
  out[i] = (_Float16)(v > 0.f ? v : 0.f);
}

// out = w*h_gat + (1-w)*(gcn_agg + bg2)   (w read on-device: capture-safe)
__global__ void k_fuse(const float* __restrict__ hgat,
                       const float* __restrict__ gcn,
                       const float* __restrict__ bg2,
                       const float* __restrict__ w, float* __restrict__ out,
                       long n) {
  long i = (long)blockIdx.x * blockDim.x + threadIdx.x;
  if (i >= n) return;
  const float ww = w[0];
  out[i] = ww * hgat[i] + (1.f - ww) * (gcn[i] + bg2[(int)(i & 63)]);
}

// ---------------------------------------------------------------------------

extern "C" void kernel_launch(void* const* d_in, const int* in_sizes, int n_in,
                              void* d_out, int out_size, void* d_ws,
                              size_t ws_size, hipStream_t stream) {
  const float* x    = (const float*)d_in[0];
  const int*   ei   = (const int*)d_in[1];
  const float* dw   = (const float*)d_in[2];
  const float* W1   = (const float*)d_in[3];
  const float* aS1  = (const float*)d_in[4];
  const float* aD1  = (const float*)d_in[5];
  // d_in[6] = b1 (added pre-BatchNorm: per-feature constant cancels exactly)
  const float* gam  = (const float*)d_in[7];
  const float* bet  = (const float*)d_in[8];
  const float* W2   = (const float*)d_in[9];
  const float* aS2  = (const float*)d_in[10];
  const float* aD2  = (const float*)d_in[11];
  const float* b2   = (const float*)d_in[12];
  const float* Wg1  = (const float*)d_in[13];
  const float* bg1  = (const float*)d_in[14];
  const float* Wg2  = (const float*)d_in[15];
  const float* bg2  = (const float*)d_in[16];
  float* out = (float*)d_out;

  const int  N    = in_sizes[0] / 128;
  const int  E    = in_sizes[1] / 2;
  const long Etot = (long)E + N;

  // ---- workspace carve (peak ~170 MB) ----
  char* wsp = (char*)d_ws;
  auto alloc = [&](size_t bytes) {
    char* r = wsp;
    wsp += (bytes + 255) & ~(size_t)255;
    return (void*)r;
  };
  _Float16* xh    = (_Float16*)alloc((size_t)N * 128 * 2);
  _Float16* W1h   = (_Float16*)alloc(128 * 256 * 2);
  _Float16* W2h   = (_Float16*)alloc(256 * 256 * 2);
  _Float16* Wg1h  = (_Float16*)alloc(128 * 64 * 2);
  _Float16* Wg2h  = (_Float16*)alloc(64 * 64 * 2);
  float*    gemmO = (float*)alloc((size_t)N * 256 * 4);
  float*    aggO  = (float*)alloc((size_t)N * 256 * 4);
  _Float16* hbn16 = (_Float16*)alloc((size_t)N * 256 * 2); // also reused as g16
  float*    sS    = (float*)alloc((size_t)N * 4 * 4);
  float*    sD    = (float*)alloc((size_t)N * 4 * 4);
  float*    zbuf  = (float*)alloc((size_t)N * 4 * 4);
  float*    pbuf  = (float*)alloc((size_t)Etot * 4 * 4);
  float*    deg   = (float*)alloc((size_t)N * 4);
  float*    hgat  = (float*)alloc((size_t)N * 64 * 4);
  float*    bnsum = (float*)alloc(256 * 4);
  float*    bnsq  = (float*)alloc(256 * 4);
  float*    bnsc  = (float*)alloc(256 * 4);
  float*    bnsh  = (float*)alloc(256 * 4);

  auto blk1 = [](long n) { return dim3((unsigned)((n + 255) / 256)); };

  // ---- f16 staging ----
  k_f32_to_f16<<<blk1((long)N * 128), 256, 0, stream>>>(x, xh, (long)N * 128);
  k_f32_to_f16<<<blk1(128 * 256), 256, 0, stream>>>(W1, W1h, 128 * 256);
  k_f32_to_f16<<<blk1(256 * 256), 256, 0, stream>>>(W2, W2h, 256 * 256);
  k_f32_to_f16<<<blk1(128 * 64), 256, 0, stream>>>(Wg1, Wg1h, 128 * 64);
  k_f32_to_f16<<<blk1(64 * 64), 256, 0, stream>>>(Wg2, Wg2h, 64 * 64);

  const dim3 gGemm256((N + 127) / 128, 8);   // Nc=256, 32 cols/block
  const dim3 gGemm64((N + 127) / 128, 2);    // Nc=64
  const dim3 gEdgeWave((unsigned)((Etot + 7) / 8));

  // ================= GAT layer 1 =================
  gemm_f16_wmma<<<gGemm256, 256, 0, stream>>>(xh, W1h, gemmO, N, 128, 256);
  k_head_dots<<<blk1((long)N * 4), 256, 0, stream>>>(gemmO, aS1, aD1, sS, sD, N);
  k_zero_f32<<<blk1((long)N * 4), 256, 0, stream>>>(zbuf, (long)N * 4);
  k_zero_f32<<<blk1((long)N * 256), 256, 0, stream>>>(aggO, (long)N * 256);
  k_edge_scores<<<blk1(Etot), 256, 0, stream>>>(ei, E, N, sS, sD, pbuf, zbuf);
  k_gat_aggregate<<<gEdgeWave, 256, 0, stream>>>(ei, E, N, pbuf, zbuf, gemmO, aggO);

  // ---- BatchNorm + ELU -> f16 ----
  k_zero_f32<<<1, 256, 0, stream>>>(bnsum, 256);
  k_zero_f32<<<1, 256, 0, stream>>>(bnsq, 256);
  k_bn_reduce<<<dim3((N + 127) / 128), 256, 0, stream>>>(aggO, bnsum, bnsq, N);
  k_bn_finalize<<<1, 256, 0, stream>>>(bnsum, bnsq, gam, bet, bnsc, bnsh,
                                       1.0f / (float)N);
  k_bn_elu_f16<<<blk1((long)N * 256), 256, 0, stream>>>(aggO, bnsc, bnsh, hbn16,
                                                        (long)N * 256);

  // ================= GAT layer 2 =================
  gemm_f16_wmma<<<gGemm256, 256, 0, stream>>>(hbn16, W2h, gemmO, N, 256, 256);
  k_head_dots<<<blk1((long)N * 4), 256, 0, stream>>>(gemmO, aS2, aD2, sS, sD, N);
  k_zero_f32<<<blk1((long)N * 4), 256, 0, stream>>>(zbuf, (long)N * 4);
  k_zero_f32<<<blk1((long)N * 256), 256, 0, stream>>>(aggO, (long)N * 256);
  k_edge_scores<<<blk1(Etot), 256, 0, stream>>>(ei, E, N, sS, sD, pbuf, zbuf);
  k_gat_aggregate<<<gEdgeWave, 256, 0, stream>>>(ei, E, N, pbuf, zbuf, gemmO, aggO);
  k_head_mean_bias<<<blk1((long)N * 64), 256, 0, stream>>>(aggO, b2, hgat, N);

  // ================= GCN branch =================
  k_deg_init<<<blk1(N), 256, 0, stream>>>(deg, N);
  k_deg_accum<<<blk1(E), 256, 0, stream>>>(ei, E, deg);
  k_dinv<<<blk1(N), 256, 0, stream>>>(deg, N);   // deg -> dinv in place

  gemm_f16_wmma<<<gGemm64, 256, 0, stream>>>(xh, Wg1h, gemmO, N, 128, 64);
  k_zero_f32<<<blk1((long)N * 64), 256, 0, stream>>>(aggO, (long)N * 64);
  k_gcn_aggregate<<<gEdgeWave, 256, 0, stream>>>(ei, E, N, deg, gemmO, aggO);
  _Float16* g16 = hbn16;  // reuse staging buffer
  k_bias_relu_f16<<<blk1((long)N * 64), 256, 0, stream>>>(aggO, bg1, g16,
                                                          (long)N * 64);

  gemm_f16_wmma<<<gGemm64, 256, 0, stream>>>(g16, Wg2h, gemmO, N, 64, 64);
  k_zero_f32<<<blk1((long)N * 64), 256, 0, stream>>>(aggO, (long)N * 64);
  k_gcn_aggregate<<<gEdgeWave, 256, 0, stream>>>(ei, E, N, deg, gemmO, aggO);

  // ================= fusion =================
  k_fuse<<<blk1((long)N * 64), 256, 0, stream>>>(hgat, aggO, bg2, dw, out,
                                                 (long)N * 64);
}
